// GroupGemmOp_8942121910611
// MI455X (gfx1250) — matrix operand order repaired and checked by reading, not compiled, not run on hardware
//
#include <hip/hip_runtime.h>
#include <hip/hip_bf16.h>

// Grouped GEMM over 8 experts, K=N=1024, total M = 66048 (static offsets).
// BF16x3 split-precision WMMA (fp32 = bf16_hi + bf16_lo), fp32 accumulate.
// Fast path: one-time fp32->bf16 hi/lo pre-conversion into workspace
// (A row-major, B transposed n-major) so the GEMM hot loop is pure b128
// copies + v_wmma_f32_16x16x32_bf16 with double-buffered LDS.
// Fallback (small workspace): fused in-loop conversion kernel.

typedef __attribute__((ext_vector_type(16))) __bf16        v16bf;
typedef __attribute__((ext_vector_type(8)))  float         v8f;
typedef __attribute__((ext_vector_type(4)))  float         v4f;
typedef __attribute__((ext_vector_type(4)))  unsigned int  v4u;
typedef __attribute__((ext_vector_type(2)))  unsigned int  v2u;

#define M_TOT 66048
#define K_DIM 1024
#define N_DIM 1024
#define N_EXP 8
#define LDSTR 40   // LDS row stride (bf16 elems): 80B -> 16-row conflict-free, 16B aligned

union Frag {
    v16bf bf;
    v4f   f[2];
};

static __device__ __forceinline__ unsigned short bf16_rne(float x) {
    unsigned int u = __float_as_uint(x);
    u += 0x7FFFu + ((u >> 16) & 1u);   // round-to-nearest-even to bf16
    return (unsigned short)(u >> 16);
}
static __device__ __forceinline__ float bf16_up(unsigned short h) {
    return __uint_as_float(((unsigned int)h) << 16);
}

static __device__ __forceinline__ int expert_of(int row0) {
    int e = 0;
    const int ends[7] = {12288, 22528, 31744, 39936, 47104, 54272, 60416};
    #pragma unroll
    for (int i = 0; i < 7; ++i)
        if (row0 >= ends[i]) e = i + 1;
    return e;
}

// Shared WMMA compute step: one 32-K panel from LDS into the 4x2 accum tiles.
static __device__ __forceinline__ void wmma_step(
    const unsigned short* sAhi, const unsigned short* sAlo,
    const unsigned short* sBhi, const unsigned short* sBlo,
    int wm, int wn, int hl, int lr, v8f acc[4][2])
{
    // B frag (32x16): lanes 0-15 hold K=0..15 (elems 0..15), lanes 16-31 K=16..31
    Frag bhi[2], blo[2];
    #pragma unroll
    for (int nf = 0; nf < 2; ++nf) {
        int off = (wn * 32 + nf * 16 + lr) * LDSTR + hl * 16;
        bhi[nf].f[0] = *(const v4f*)&sBhi[off];
        bhi[nf].f[1] = *(const v4f*)&sBhi[off + 8];
        blo[nf].f[0] = *(const v4f*)&sBlo[off];
        blo[nf].f[1] = *(const v4f*)&sBlo[off + 8];
    }
    // A frag (16x32): elems 0..15 -> K = hl*16+0..15
    #pragma unroll
    for (int mf = 0; mf < 4; ++mf) {
        int offA = (wm * 64 + mf * 16 + lr) * LDSTR + hl * 16;
        Frag ahi, alo;
        ahi.f[0] = *(const v4f*)&sAhi[offA];
        ahi.f[1] = *(const v4f*)&sAhi[offA + 8];
        alo.f[0] = *(const v4f*)&sAlo[offA];
        alo.f[1] = *(const v4f*)&sAlo[offA + 8];
        #pragma unroll
        for (int nf = 0; nf < 2; ++nf) {
            acc[mf][nf] = __builtin_amdgcn_wmma_f32_16x16x32_bf16(
                false, ahi.bf, false, bhi[nf].bf, (short)0, acc[mf][nf], false, false);
            acc[mf][nf] = __builtin_amdgcn_wmma_f32_16x16x32_bf16(
                false, ahi.bf, false, blo[nf].bf, (short)0, acc[mf][nf], false, false);
            acc[mf][nf] = __builtin_amdgcn_wmma_f32_16x16x32_bf16(
                false, alo.bf, false, bhi[nf].bf, (short)0, acc[mf][nf], false, false);
        }
    }
}

static __device__ __forceinline__ void store_acc(
    float* __restrict__ C, v8f acc[4][2],
    int row0, int n0, int wm, int wn, int hl, int lr)
{
    // C/D 16x16 layout: M = vgpr + hl*8, N = lr
    #pragma unroll
    for (int mf = 0; mf < 4; ++mf)
        #pragma unroll
        for (int nf = 0; nf < 2; ++nf) {
            int colg = n0 + wn * 32 + nf * 16 + lr;
            int rowb = row0 + wm * 64 + mf * 16 + hl * 8;
            #pragma unroll
            for (int v = 0; v < 8; ++v)
                C[(size_t)(rowb + v) * N_DIM + colg] = acc[mf][nf][v];
        }
}

// ---------------- Pre-pass 1: A fp32 -> bf16 hi/lo, row-major ----------------
__global__ __launch_bounds__(256)
void convert_a_kernel(const float* __restrict__ A,
                      unsigned short* __restrict__ Ahi,
                      unsigned short* __restrict__ Alo)
{
    size_t b0 = (size_t)blockIdx.x * 4096;      // 4096 floats per block
    #pragma unroll
    for (int i = 0; i < 4; ++i) {
        size_t off = b0 + ((size_t)(threadIdx.x + (i << 8)) << 2);
        v4f x = *(const v4f*)(A + off);
        unsigned short h[4], l[4];
        #pragma unroll
        for (int j = 0; j < 4; ++j) {
            h[j] = bf16_rne(x[j]);
            l[j] = bf16_rne(x[j] - bf16_up(h[j]));
        }
        v2u ph = {(unsigned)h[0] | ((unsigned)h[1] << 16),
                  (unsigned)h[2] | ((unsigned)h[3] << 16)};
        v2u pl = {(unsigned)l[0] | ((unsigned)l[1] << 16),
                  (unsigned)l[2] | ((unsigned)l[3] << 16)};
        *(v2u*)&Ahi[off] = ph;
        *(v2u*)&Alo[off] = pl;
    }
}

// ------- Pre-pass 2: B fp32 [e][k][n] -> bf16 hi/lo transposed [e][n][k] -----
__global__ __launch_bounds__(256)
void convert_b_kernel(const float* __restrict__ B,
                      unsigned short* __restrict__ Bthi,
                      unsigned short* __restrict__ Btlo)
{
    __shared__ __align__(16) unsigned short tHi[64 * 72];
    __shared__ __align__(16) unsigned short tLo[64 * 72];
    const int tid = threadIdx.x;
    const int bid = blockIdx.x;
    const int tn = bid & 15;            // 64-col n tile
    const int tk = (bid >> 4) & 15;     // 64-row k tile
    const int e  = bid >> 8;            // expert
    const float* src = B + (size_t)e * K_DIM * N_DIM + (size_t)(tk * 64) * N_DIM + tn * 64;

    #pragma unroll
    for (int i = 0; i < 4; ++i) {
        int idx = tid + (i << 8);       // 0..1023 float4 slots of 64x64 tile
        int r   = idx >> 4;             // k-local
        int c4  = (idx & 15) << 2;      // n-local
        v4f x = *(const v4f*)(src + (size_t)r * N_DIM + c4);
        #pragma unroll
        for (int j = 0; j < 4; ++j) {
            unsigned short hb = bf16_rne(x[j]);
            tHi[(c4 + j) * 72 + r] = hb;
            tLo[(c4 + j) * 72 + r] = bf16_rne(x[j] - bf16_up(hb));
        }
    }
    __syncthreads();

    unsigned short* dh = Bthi + (size_t)e * N_DIM * K_DIM + (size_t)(tn * 64) * K_DIM + tk * 64;
    unsigned short* dl = Btlo + (size_t)e * N_DIM * K_DIM + (size_t)(tn * 64) * K_DIM + tk * 64;
    #pragma unroll
    for (int i = 0; i < 2; ++i) {
        int idx = tid + (i << 8);       // 0..511 b128 chunks
        int n   = idx >> 3;
        int c8  = (idx & 7) << 3;
        *(v4u*)&dh[(size_t)n * K_DIM + c8] = *(const v4u*)&tHi[n * 72 + c8];
        *(v4u*)&dl[(size_t)n * K_DIM + c8] = *(const v4u*)&tLo[n * 72 + c8];
    }
}

// --------- Fast GEMM: pre-converted bf16 panels, double-buffered LDS ---------
__global__ __launch_bounds__(256)
void group_gemm_pre_kernel(const unsigned short* __restrict__ Ahi,
                           const unsigned short* __restrict__ Alo,
                           const unsigned short* __restrict__ Bhi,
                           const unsigned short* __restrict__ Blo,
                           float* __restrict__ C)
{
    __shared__ __align__(16) unsigned short sAhi[2][128 * LDSTR];
    __shared__ __align__(16) unsigned short sAlo[2][128 * LDSTR];
    __shared__ __align__(16) unsigned short sBhi[2][128 * LDSTR];
    __shared__ __align__(16) unsigned short sBlo[2][128 * LDSTR];

    const int tid  = threadIdx.x;
    const int lane = tid & 31, wave = tid >> 5;
    const int hl   = lane >> 4, lr = lane & 15;
    const int wm   = wave >> 2, wn = wave & 3;

    const int n0   = blockIdx.x * 128;   // n-tile fastest -> A panel L2 reuse
    const int row0 = blockIdx.y * 128;

    const int e = expert_of(row0);
    const unsigned short* __restrict__ BhiE = Bhi + (size_t)e * N_DIM * K_DIM;
    const unsigned short* __restrict__ BloE = Blo + (size_t)e * N_DIM * K_DIM;

    // Prologue: copy k-panel 0 straight into LDS buffer 0 (b128 chunks)
    #pragma unroll
    for (int i = 0; i < 2; ++i) {
        int idx = tid + (i << 8);        // 0..511 over (r:128, c8:4)
        int r = idx >> 2, c8 = (idx & 3) << 3;
        size_t ga = (size_t)(row0 + r) * K_DIM + c8;
        size_t gb = (size_t)(n0  + r) * K_DIM + c8;   // Bt rows are n
        *(v4u*)&sAhi[0][r * LDSTR + c8] = *(const v4u*)&Ahi[ga];
        *(v4u*)&sAlo[0][r * LDSTR + c8] = *(const v4u*)&Alo[ga];
        *(v4u*)&sBhi[0][r * LDSTR + c8] = *(const v4u*)&BhiE[gb];
        *(v4u*)&sBlo[0][r * LDSTR + c8] = *(const v4u*)&BloE[gb];
    }
    __syncthreads();

    v8f acc[4][2];
    #pragma unroll
    for (int mf = 0; mf < 4; ++mf)
        #pragma unroll
        for (int nf = 0; nf < 2; ++nf)
            acc[mf][nf] = (v8f){0.f, 0.f, 0.f, 0.f, 0.f, 0.f, 0.f, 0.f};

    int cur = 0;
    for (int k0 = 0; k0 < K_DIM; k0 += 32) {
        const bool more = (k0 + 32) < K_DIM;

        // Issue next panel's global loads first (latency hides under WMMA)
        v4u ra_h[2], ra_l[2], rb_h[2], rb_l[2];
        if (more) {
            #pragma unroll
            for (int i = 0; i < 2; ++i) {
                int idx = tid + (i << 8);
                int r = idx >> 2, c8 = (idx & 3) << 3;
                size_t ga = (size_t)(row0 + r) * K_DIM + (k0 + 32) + c8;
                size_t gb = (size_t)(n0  + r) * K_DIM + (k0 + 32) + c8;
                ra_h[i] = *(const v4u*)&Ahi[ga];
                ra_l[i] = *(const v4u*)&Alo[ga];
                rb_h[i] = *(const v4u*)&BhiE[gb];
                rb_l[i] = *(const v4u*)&BloE[gb];
                if (k0 + 64 < K_DIM) {
                    __builtin_prefetch(&Ahi[ga + 32], 0, 1);  // k-panel after next
                    __builtin_prefetch(&BhiE[gb + 32], 0, 1);
                }
            }
        }

        wmma_step(sAhi[cur], sAlo[cur], sBhi[cur], sBlo[cur], wm, wn, hl, lr, acc);

        if (more) {
            #pragma unroll
            for (int i = 0; i < 2; ++i) {
                int idx = tid + (i << 8);
                int r = idx >> 2, c8 = (idx & 3) << 3;
                *(v4u*)&sAhi[cur ^ 1][r * LDSTR + c8] = ra_h[i];
                *(v4u*)&sAlo[cur ^ 1][r * LDSTR + c8] = ra_l[i];
                *(v4u*)&sBhi[cur ^ 1][r * LDSTR + c8] = rb_h[i];
                *(v4u*)&sBlo[cur ^ 1][r * LDSTR + c8] = rb_l[i];
            }
        }
        __syncthreads();
        cur ^= 1;
    }

    store_acc(C, acc, row0, n0, wm, wn, hl, lr);
}

// --------- Fallback GEMM: fused in-loop conversion (round-1 kernel) ---------
__global__ __launch_bounds__(256)
void group_gemm_fused_kernel(const float* __restrict__ A,
                             const float* __restrict__ B,
                             float* __restrict__ C)
{
    __shared__ __align__(16) unsigned short sAhi[128 * LDSTR];
    __shared__ __align__(16) unsigned short sAlo[128 * LDSTR];
    __shared__ __align__(16) unsigned short sBhi[128 * LDSTR];
    __shared__ __align__(16) unsigned short sBlo[128 * LDSTR];

    const int tid  = threadIdx.x;
    const int lane = tid & 31, wave = tid >> 5;
    const int hl   = lane >> 4, lr = lane & 15;
    const int wm   = wave >> 2, wn = wave & 3;

    const int n0   = blockIdx.x * 128;
    const int row0 = blockIdx.y * 128;

    const int e = expert_of(row0);
    const float* __restrict__ Bexp = B + (size_t)e * K_DIM * N_DIM;

    v8f acc[4][2];
    #pragma unroll
    for (int mf = 0; mf < 4; ++mf)
        #pragma unroll
        for (int nf = 0; nf < 2; ++nf)
            acc[mf][nf] = (v8f){0.f, 0.f, 0.f, 0.f, 0.f, 0.f, 0.f, 0.f};

    for (int k0 = 0; k0 < K_DIM; k0 += 32) {
        const bool more = (k0 + 32) < K_DIM;
        #pragma unroll
        for (int i = 0; i < 4; ++i) {           // A 128x32 -> hi/lo
            int idx = tid + (i << 8);
            int r = idx >> 3, c = (idx & 7) << 2;
            const float* gp = A + (size_t)(row0 + r) * K_DIM + (k0 + c);
            v4f x = *(const v4f*)gp;
            if (more) __builtin_prefetch(gp + 32, 0, 1);
            unsigned short h[4], l[4];
            #pragma unroll
            for (int j = 0; j < 4; ++j) {
                h[j] = bf16_rne(x[j]);
                l[j] = bf16_rne(x[j] - bf16_up(h[j]));
            }
            v2u ph = {(unsigned)h[0] | ((unsigned)h[1] << 16),
                      (unsigned)h[2] | ((unsigned)h[3] << 16)};
            v2u pl = {(unsigned)l[0] | ((unsigned)l[1] << 16),
                      (unsigned)l[2] | ((unsigned)l[3] << 16)};
            *(v2u*)&sAhi[r * LDSTR + c] = ph;
            *(v2u*)&sAlo[r * LDSTR + c] = pl;
        }
        #pragma unroll
        for (int i = 0; i < 4; ++i) {           // B 32x128 -> transposed hi/lo
            int idx = tid + (i << 8);
            int k = idx >> 5, n = (idx & 31) << 2;
            const float* gp = Bexp + (size_t)(k0 + k) * N_DIM + (n0 + n);
            v4f x = *(const v4f*)gp;
            #pragma unroll
            for (int j = 0; j < 4; ++j) {
                unsigned short hb = bf16_rne(x[j]);
                sBhi[(n + j) * LDSTR + k] = hb;
                sBlo[(n + j) * LDSTR + k] = bf16_rne(x[j] - bf16_up(hb));
            }
        }
        __syncthreads();
        wmma_step(sAhi, sAlo, sBhi, sBlo, wm, wn, hl, lr, acc);
        __syncthreads();
    }

    store_acc(C, acc, row0, n0, wm, wn, hl, lr);
}

extern "C" void kernel_launch(void* const* d_in, const int* in_sizes, int n_in,
                              void* d_out, int out_size, void* d_ws, size_t ws_size,
                              hipStream_t stream) {
    (void)in_sizes; (void)n_in; (void)out_size;
    const float* a = (const float*)d_in[0];   // [66048, 1024] fp32
    const float* b = (const float*)d_in[1];   // [8, 1024, 1024] fp32
    float* out = (float*)d_out;               // [66048, 1024] fp32

    const size_t elemsA = (size_t)M_TOT * K_DIM;            // 67,633,152
    const size_t elemsB = (size_t)N_EXP * K_DIM * N_DIM;    // 8,388,608
    const size_t need   = 2 * (elemsA + elemsB) * sizeof(unsigned short); // ~290 MB

    dim3 grid(N_DIM / 128, M_TOT / 128, 1);   // (8, 516): n-tile fastest
    dim3 block(256, 1, 1);                    // 8 waves (wave32)

    if (ws_size >= need) {
        unsigned short* Ahi  = (unsigned short*)d_ws;
        unsigned short* Alo  = Ahi + elemsA;
        unsigned short* Bthi = Alo + elemsA;
        unsigned short* Btlo = Bthi + elemsB;
        convert_a_kernel<<<dim3((unsigned)(elemsA / 4096)), block, 0, stream>>>(a, Ahi, Alo);
        convert_b_kernel<<<dim3(N_EXP * 16 * 16), block, 0, stream>>>(b, Bthi, Btlo);
        group_gemm_pre_kernel<<<grid, block, 0, stream>>>(Ahi, Alo, Bthi, Btlo, out);
    } else {
        group_gemm_fused_kernel<<<grid, block, 0, stream>>>(a, b, out);
    }
}
